// MultiHeadAttention_45397804318865
// MI455X (gfx1250) — matrix operand order repaired and checked by reading, not compiled, not run on hardware
//
#include <hip/hip_runtime.h>
#include <hip/hip_bf16.h>

// Problem constants (from reference): B=4, S=2048, D=1024, H=16, DK=64
#define PB   4
#define PS   2048
#define PD   1024
#define PH   16
#define PDK  64
#define PM   (PB * PS)       // 8192 rows for the projection GEMMs

typedef __attribute__((ext_vector_type(16))) _Float16 v16h;
typedef __attribute__((ext_vector_type(8)))  _Float16 v8h;
typedef __attribute__((ext_vector_type(2)))  __fp16   v2hf;
typedef __attribute__((ext_vector_type(4)))  __fp16   v4hf;
typedef __attribute__((ext_vector_type(8)))  __fp16   v8hf;
typedef __attribute__((ext_vector_type(8)))  float    v8f;
typedef __attribute__((ext_vector_type(4)))  unsigned v4u;
typedef __attribute__((ext_vector_type(8)))  int      v8i;
typedef __attribute__((ext_vector_type(4)))  int      v4i;

__device__ __forceinline__ v8f wmma_f16(v16h a, v16h b, v8f c) {
  // (neg_a, A, neg_b, B, c_mod, C, reuse_a, reuse_b)
  return __builtin_amdgcn_wmma_f32_16x16x32_f16(false, a, false, b, (short)0, c,
                                                false, false);
}

// Pack 8 f32 -> 8 f16 with v_cvt_pk_rtz_f16_f32 (2 elems / VALU op)
__device__ __forceinline__ v8h pack8(float4 a, float4 b) {
  v2hf p0 = __builtin_amdgcn_cvt_pkrtz(a.x, a.y);
  v2hf p1 = __builtin_amdgcn_cvt_pkrtz(a.z, a.w);
  v2hf p2 = __builtin_amdgcn_cvt_pkrtz(b.x, b.y);
  v2hf p3 = __builtin_amdgcn_cvt_pkrtz(b.z, b.w);
  v4hf q0 = __builtin_shufflevector(p0, p1, 0, 1, 2, 3);
  v4hf q1 = __builtin_shufflevector(p2, p3, 0, 1, 2, 3);
  v8hf r  = __builtin_shufflevector(q0, q1, 0, 1, 2, 3, 4, 5, 6, 7);
  return __builtin_bit_cast(v8h, r);
}

// ---- A-operand fragment (16x32 f16, M x K) from an LDS tile [row][k], ld halves
// ISA layout: lanes 0-15 hold M=lane, K = {0..7, 16..23}; lanes 16-31 hold
// M=lane-16, K = {8..15, 24..31}.
__device__ __forceinline__ v16h frag_a(const _Float16* tile, int ld) {
  const int lane = threadIdx.x & 31;
  const int r    = lane & 15;
  const int kb   = (lane >> 4) << 3;     // 0 or 8
  const _Float16* p = tile + r * ld + kb;
  v8h lo = *(const v8h*)(p);             // K = kb .. kb+7
  v8h hi = *(const v8h*)(p + 16);        // K = kb+16 .. kb+23
  return __builtin_shufflevector(lo, hi, 0,1,2,3,4,5,6,7,8,9,10,11,12,13,14,15);
}

// ---- B-operand fragment (32x16 f16, K x N) from an LDS tile stored [n][k]
// ISA layout: lane holds column N=lane&15; lanes 0-15 hold K = {0..7, 16..23},
// lanes 16-31 hold K = {8..15, 24..31} (same K order as frag_a).
__device__ __forceinline__ v16h frag_b(const _Float16* tile, int ld) {
  const int lane = threadIdx.x & 31;
  const int n    = lane & 15;
  const int kb   = (lane >> 4) << 3;     // 0 or 8
  const _Float16* p = tile + n * ld + kb;
  v8h lo = *(const v8h*)(p);
  v8h hi = *(const v8h*)(p + 16);
  return __builtin_shufflevector(lo, hi, 0,1,2,3,4,5,6,7,8,9,10,11,12,13,14,15);
}

// ---- Tensor Data Mover: async-load a 64x64 f16 tile (contiguous rows of a
// [2048 x 64] f16 tensor) into LDS.  D# per CDNA5 ISA ch.8: group0 holds
// count/lds_addr/global_addr/type, group1 holds data_size, tensor dims 2048x64,
// tile 64x64, dim0_stride 64, dim1_stride 64*2048.  Tracked by TENSORcnt.
__device__ __forceinline__ void tdm_load_64x64_f16(unsigned lds_addr,
                                                   unsigned long long gaddr) {
  v4u g0;
  g0.x = 1u;                                              // count = 1 descriptor
  g0.y = lds_addr;                                        // LDS byte address
  g0.z = (unsigned)gaddr;                                 // global_addr[31:0]
  g0.w = ((unsigned)(gaddr >> 32) & 0x01FFFFFFu) | (2u << 30);  // [56:32]|type=2
  v8i g1;
  g1[0] = (int)(1u << 16);            // workgroup_mask=0, data_size=1 (2 bytes)
  g1[1] = (int)(64u << 16);           // tensor_dim0 = 64     (bits 79:48 lo)
  g1[2] = (int)(2048u << 16);         // tensor_dim1 = 2048   (bits 111:80 lo)
  g1[3] = (int)(64u << 16);           // tile_dim0 = 64       (bits 127:112)
  g1[4] = 64;                         // tile_dim1 = 64       (bits 143:128)
  g1[5] = 64;                         // tensor_dim0_stride = 64 (bits 207:160)
  g1[6] = 0;                          // dim0_stride hi, dim1_stride lo16 (=0)
  g1[7] = 2;                          // tensor_dim1_stride = 131072 (hi32 = 2)
  v4i z4 = {0, 0, 0, 0};
  v8i z8 = {0, 0, 0, 0, 0, 0, 0, 0};
  __builtin_amdgcn_tensor_load_to_lds(g0, g1, z4, z4, z8, 0);
}

// ============================================================================
// GEMM: Y = X @ W^T + bias.  X: [M,K] (f32 or f16), W: [N,K] f32, bias: [N].
// OUT_MODE 0: Y f32 row-major [M,N] (final projection -> d_out)
// OUT_MODE 1: Y f16 head-split [B,H,S,DK] (Q/K/V projections)
// Block tile 128(M) x 128(N), K-step 32, 256 threads = 8 waves, wave tile 32x64.
// Double-buffered LDS: global loads of tile k+1 overlap the 8 WMMAs of tile k.
// ============================================================================
#define GBM 128
#define GBN 128
#define GBK 32
#define GLD 40   // GBK + 8 halves; rows stay 16B-aligned (80B per row)

template <bool A_HALF, int OUT_MODE>
__global__ __launch_bounds__(256) void gemm_wmma_kernel(
    const void* __restrict__ Xv, const float* __restrict__ W,
    const float* __restrict__ bias, void* __restrict__ Yv,
    int M, int N, int K) {
  __shared__ __align__(16) _Float16 As[2][GBM][GLD];
  __shared__ __align__(16) _Float16 Bs[2][GBN][GLD];

  const int tid  = threadIdx.x;
  const int lane = tid & 31;
  const int wave = tid >> 5;
  const int wm   = (wave >> 1) * 32;         // 0,32,64,96
  const int wn   = (wave & 1) * 64;          // 0,64
  const int m0   = blockIdx.x * GBM;
  const int n0   = blockIdx.y * GBN;

  const float*    Xf = (const float*)Xv;
  const _Float16* Xh = (const _Float16*)Xv;

  // cooperative-load mapping: 128 rows x 32 K, 16 elems per thread
  const int ar = tid >> 1, ac = (tid & 1) * 16;
  const int br = tid >> 1, bc = (tid & 1) * 16;

  v8f acc[2][4] = {};

  // ---- prologue: stage K-tile 0 into buffer 0
  if (A_HALF) {
    const v8h* s = (const v8h*)(Xh + (size_t)(m0 + ar) * K + ac);
    *(v8h*)&As[0][ar][ac]     = s[0];
    *(v8h*)&As[0][ar][ac + 8] = s[1];
  } else {
    const float4* s = (const float4*)(Xf + (size_t)(m0 + ar) * K + ac);
    *(v8h*)&As[0][ar][ac]     = pack8(s[0], s[1]);
    *(v8h*)&As[0][ar][ac + 8] = pack8(s[2], s[3]);
  }
  {
    const float4* s = (const float4*)(W + (size_t)(n0 + br) * K + bc);
    *(v8h*)&Bs[0][br][bc]     = pack8(s[0], s[1]);
    *(v8h*)&Bs[0][br][bc + 8] = pack8(s[2], s[3]);
  }

  int cur = 0;
  for (int kt = 0; kt < K; kt += GBK) {
    const bool has_next = (kt + GBK) < K;
    float4 ra[4], rb[4];
    v8h rah[2];
    if (has_next) {
      if (A_HALF) {
        const v8h* s = (const v8h*)(Xh + (size_t)(m0 + ar) * K + kt + GBK + ac);
        rah[0] = s[0]; rah[1] = s[1];
      } else {
        const float4* s =
            (const float4*)(Xf + (size_t)(m0 + ar) * K + kt + GBK + ac);
        ra[0] = s[0]; ra[1] = s[1]; ra[2] = s[2]; ra[3] = s[3];
      }
      const float4* s = (const float4*)(W + (size_t)(n0 + br) * K + kt + GBK + bc);
      rb[0] = s[0]; rb[1] = s[1]; rb[2] = s[2]; rb[3] = s[3];
      if (kt + 2 * GBK < K) {   // keep the cache one tile ahead of the pipeline
        __builtin_prefetch(W + (size_t)(n0 + br) * K + kt + 2 * GBK + bc, 0, 1);
        if (A_HALF)
          __builtin_prefetch(Xh + (size_t)(m0 + ar) * K + kt + 2 * GBK + ac, 0, 1);
        else
          __builtin_prefetch(Xf + (size_t)(m0 + ar) * K + kt + 2 * GBK + ac, 0, 1);
      }
    }
    __syncthreads();   // LDS[cur] stores visible; previous readers of nxt done

    v16h a0 = frag_a(&As[cur][wm][0], GLD);
    v16h a1 = frag_a(&As[cur][wm + 16][0], GLD);
#pragma unroll
    for (int tn = 0; tn < 4; ++tn) {
      v16h b = frag_b(&Bs[cur][wn + tn * 16][0], GLD);
      acc[0][tn] = wmma_f16(a0, b, acc[0][tn]);
      acc[1][tn] = wmma_f16(a1, b, acc[1][tn]);
    }

    if (has_next) {
      const int nxt = cur ^ 1;
      if (A_HALF) {
        *(v8h*)&As[nxt][ar][ac]     = rah[0];
        *(v8h*)&As[nxt][ar][ac + 8] = rah[1];
      } else {
        *(v8h*)&As[nxt][ar][ac]     = pack8(ra[0], ra[1]);
        *(v8h*)&As[nxt][ar][ac + 8] = pack8(ra[2], ra[3]);
      }
      *(v8h*)&Bs[nxt][br][bc]     = pack8(rb[0], rb[1]);
      *(v8h*)&Bs[nxt][br][bc + 8] = pack8(rb[2], rb[3]);
      cur = nxt;
    }
  }

  // ---- epilogue: C layout = VGPR i -> row i + 8*(lane>=16), col = lane&15
  const int col   = lane & 15;
  const int rhalf = (lane >> 4) * 8;
#pragma unroll
  for (int tm = 0; tm < 2; ++tm) {
#pragma unroll
    for (int tn = 0; tn < 4; ++tn) {
      const int n  = n0 + wn + tn * 16 + col;
      const float bv = bias[n];
#pragma unroll
      for (int i = 0; i < 8; ++i) {
        const int m = m0 + wm + tm * 16 + rhalf + i;
        const float y = acc[tm][tn][i] + bv;
        if (OUT_MODE == 0) {
          ((float*)Yv)[(size_t)m * N + n] = y;
        } else {
          const int b = m / PS, s = m % PS;
          const int h = n / PDK, dk = n % PDK;
          ((_Float16*)Yv)[((((size_t)b * PH + h) * PS) + s) * PDK + dk] =
              (_Float16)y;
        }
      }
    }
  }
}

// ============================================================================
// Flash attention: causal softmax(Q K^T / sqrt(DK)) V over f16 Q/K/V in
// [B,H,S,DK] layout.  64 queries / block, 4 waves x 16 rows, key tiles of 64.
// Q and K tiles staged by the Tensor Data Mover; base-2 online softmax.
// ============================================================================
#define FLD 72   // padded ld for Vs / Ps (rows 16B-aligned: 144B)
#define KLD 64   // unpadded ld for TDM-written tiles (rows 128B)

__global__ __launch_bounds__(128) void flash_attn_kernel(
    const _Float16* __restrict__ Qh, const _Float16* __restrict__ Kh,
    const _Float16* __restrict__ Vh, _Float16* __restrict__ ctx) {
  __shared__ __align__(16) _Float16 Qs[64][KLD];      // [q][d]   (TDM target)
  __shared__ __align__(16) _Float16 Ks[64][KLD];      // [kk][d]  (TDM target)
  __shared__ __align__(16) _Float16 Vs[64][FLD];      // transposed: [d][kk]
  __shared__ __align__(16) _Float16 Ps[4][16][FLD];   // per-wave P tile [m][kk]

  const int tid  = threadIdx.x;
  const int lane = tid & 31;
  const int wave = tid >> 5;
  const int bh   = blockIdx.y;          // b*H + h
  const int b    = bh >> 4;             // H == 16
  const int h    = bh & 15;
  const int qt   = blockIdx.x;
  const int q0   = qt * 64;

  const size_t head = (size_t)bh * PS * PDK;
  const _Float16* Qb = Qh + head;
  const _Float16* Kb = Kh + head;
  const _Float16* Vb = Vh + head;

  // ---- stage Q tile once via TDM
  if (wave == 0 && lane == 0) {
    tdm_load_64x64_f16((unsigned)(uintptr_t)&Qs[0][0],
                       (unsigned long long)(const void*)(Qb + (size_t)q0 * PDK));
  }
  if (wave == 0) __builtin_amdgcn_s_wait_tensorcnt(0);
  __syncthreads();
  const v16h qa0 = frag_a(&Qs[wave * 16][0], KLD);
  const v16h qa1 = frag_a(&Qs[wave * 16][32], KLD);

  float mrow[8], lrow[8];
#pragma unroll
  for (int i = 0; i < 8; ++i) { mrow[i] = -1e30f; lrow[i] = 0.f; }
  v8f o[4] = {};

  const int col   = lane & 15;
  const int rhalf = (lane >> 4) * 8;
  // base-2 softmax: fold 1/sqrt(DK) and log2(e) into one score scale
  const float kSc = 0.125f * 1.44269504088896340736f;

  for (int kt = 0; kt <= qt; ++kt) {
    __syncthreads();                    // previous readers of Ks/Vs are done
    // ---- async K tile via Tensor Data Mover (one issue per block)
    if (wave == 0 && lane == 0) {
      tdm_load_64x64_f16(
          (unsigned)(uintptr_t)&Ks[0][0],
          (unsigned long long)(const void*)(Kb + (size_t)(kt * 64) * PDK));
    }
    // ---- V tile: load + transpose into Vs (overlaps the TDM transfer)
    {
      const int r = tid >> 1, c0 = (tid & 1) * 32;
      const size_t off = (size_t)(kt * 64 + r) * PDK + c0;
      const _Float16* vsrc = Vb + off;
#pragma unroll
      for (int j = 0; j < 32; ++j) Vs[c0 + j][r] = vsrc[j];   // transpose
      if (kt < qt)                      // prefetch next V tile
        __builtin_prefetch(Vb + off + (size_t)64 * PDK, 0, 1);
    }
    if (wave == 0) __builtin_amdgcn_s_wait_tensorcnt(0);
    __syncthreads();

    // ---- scores: 16x64 per wave = 4 N-tiles x 2 K-steps of WMMA
    v8f sc[4];
#pragma unroll
    for (int t = 0; t < 4; ++t) {
      v8f s = {};
      s = wmma_f16(qa0, frag_b(&Ks[t * 16][0], KLD), s);
      s = wmma_f16(qa1, frag_b(&Ks[t * 16][32], KLD), s);
      sc[t] = s;
    }

    // ---- causal mask (analytic) + running row max, base-2 domain
    float rm[8];
#pragma unroll
    for (int i = 0; i < 8; ++i) rm[i] = -1e30f;
#pragma unroll
    for (int t = 0; t < 4; ++t) {
#pragma unroll
      for (int i = 0; i < 8; ++i) {
        const int qg = q0 + wave * 16 + rhalf + i;
        const int kg = kt * 64 + t * 16 + col;
        float v = sc[t][i] * kSc;
        v = (kg <= qg) ? v : -1e30f;
        sc[t][i] = v;
        rm[i] = fmaxf(rm[i], v);
      }
    }
    // reduce over the 16 lanes holding each row (xor stays within half-wave)
#pragma unroll
    for (int i = 0; i < 8; ++i) {
      float v = rm[i];
      v = fmaxf(v, __shfl_xor(v, 1, 32));
      v = fmaxf(v, __shfl_xor(v, 2, 32));
      v = fmaxf(v, __shfl_xor(v, 4, 32));
      v = fmaxf(v, __shfl_xor(v, 8, 32));
      rm[i] = v;
    }
    float alpha[8];
#pragma unroll
    for (int i = 0; i < 8; ++i) {
      const float mn = fmaxf(mrow[i], rm[i]);
      alpha[i] = __builtin_amdgcn_exp2f(mrow[i] - mn);
      mrow[i] = mn;
    }
    // ---- P = 2^(S - m), stage to LDS in A-operand tile layout
    float rs[8];
#pragma unroll
    for (int i = 0; i < 8; ++i) rs[i] = 0.f;
#pragma unroll
    for (int t = 0; t < 4; ++t) {
#pragma unroll
      for (int i = 0; i < 8; ++i) {
        const float p = __builtin_amdgcn_exp2f(sc[t][i] - mrow[i]);
        rs[i] += p;
        Ps[wave][rhalf + i][t * 16 + col] = (_Float16)p;
      }
    }
#pragma unroll
    for (int i = 0; i < 8; ++i) {
      float v = rs[i];
      v += __shfl_xor(v, 1, 32);
      v += __shfl_xor(v, 2, 32);
      v += __shfl_xor(v, 4, 32);
      v += __shfl_xor(v, 8, 32);
      lrow[i] = lrow[i] * alpha[i] + v;
    }
    // rescale running output
#pragma unroll
    for (int t = 0; t < 4; ++t)
#pragma unroll
      for (int i = 0; i < 8; ++i) o[t][i] *= alpha[i];

    __syncthreads();                    // make Ps visible to this wave's loads

    // ---- O += P (16x64) x V (64x64): 4 D-tiles x 2 K-steps of WMMA
    const v16h pa0 = frag_a(&Ps[wave][0][0], FLD);
    const v16h pa1 = frag_a(&Ps[wave][0][32], FLD);
#pragma unroll
    for (int t = 0; t < 4; ++t) {
      o[t] = wmma_f16(pa0, frag_b(&Vs[t * 16][0], FLD), o[t]);
      o[t] = wmma_f16(pa1, frag_b(&Vs[t * 16][32], FLD), o[t]);
    }
  }

  // ---- epilogue: O /= l, write context as [B,S,H*DK] f16
#pragma unroll
  for (int i = 0; i < 8; ++i) lrow[i] = __builtin_amdgcn_rcpf(lrow[i]);
#pragma unroll
  for (int t = 0; t < 4; ++t) {
#pragma unroll
    for (int i = 0; i < 8; ++i) {
      const int qg = q0 + wave * 16 + rhalf + i;
      const int d  = t * 16 + col;
      ctx[((size_t)b * PS + qg) * PD + h * PDK + d] =
          (_Float16)(o[t][i] * lrow[i]);
    }
  }
}

// ============================================================================
extern "C" void kernel_launch(void* const* d_in, const int* in_sizes, int n_in,
                              void* d_out, int out_size, void* d_ws,
                              size_t ws_size, hipStream_t stream) {
  (void)in_sizes; (void)n_in; (void)out_size; (void)ws_size;
  const float* q  = (const float*)d_in[0];
  const float* k  = (const float*)d_in[1];
  const float* v  = (const float*)d_in[2];
  const float* Wq = (const float*)d_in[3];
  const float* bq = (const float*)d_in[4];
  const float* Wk = (const float*)d_in[5];
  const float* bk = (const float*)d_in[6];
  const float* Wv = (const float*)d_in[7];
  const float* bv = (const float*)d_in[8];
  const float* Wo = (const float*)d_in[9];
  const float* bo = (const float*)d_in[10];
  // d_in[11] = causal mask: handled analytically

  char* ws = (char*)d_ws;
  _Float16* Qh  = (_Float16*)(ws);                        // 16 MB
  _Float16* Kh  = (_Float16*)(ws + (size_t)(16u << 20));  // 16 MB
  _Float16* Vh  = (_Float16*)(ws + (size_t)(32u << 20));  // 16 MB
  _Float16* ctx = (_Float16*)(ws + (size_t)(48u << 20));  // 16 MB

  const int M = PM, N = PD, K = PD;
  dim3 ggrid(M / GBM, N / GBN);   // (64, 8)
  gemm_wmma_kernel<false, 1><<<ggrid, 256, 0, stream>>>(q, Wq, bq, Qh, M, N, K);
  gemm_wmma_kernel<false, 1><<<ggrid, 256, 0, stream>>>(k, Wk, bk, Kh, M, N, K);
  gemm_wmma_kernel<false, 1><<<ggrid, 256, 0, stream>>>(v, Wv, bv, Vh, M, N, K);

  dim3 fgrid(PS / 64, PB * PH);   // (32, 64)
  flash_attn_kernel<<<fgrid, 128, 0, stream>>>(Qh, Kh, Vh, ctx);

  gemm_wmma_kernel<true, 0><<<ggrid, 256, 0, stream>>>(ctx, Wo, bo, d_out, M, N, K);
}
